// PixelEffectModule_7971459301886
// MI455X (gfx1250) — matrix operand order repaired and checked by reading, not compiled, not run on hardware
//
#include <hip/hip_runtime.h>

// CDNA5 WMMA vector types
typedef __attribute__((ext_vector_type(16))) _Float16     v16h;
typedef __attribute__((ext_vector_type(4)))  _Float16     v4h;
typedef __attribute__((ext_vector_type(8)))  float        v8f;
typedef __attribute__((ext_vector_type(8)))  unsigned int v8u;

// Fixed parameters from setup_inputs()
constexpr int NB = 16;            // param_num_bins
constexpr int KS = 11;            // param_kernel_size
constexpr int PS = 10;            // param_pixel_size (conv stride)
constexpr int PAD = (KS - 1) / 2; // 5
constexpr int H = 2048, W = 2048;
constexpr int HO = (H + 2 * PAD - KS) / PS + 1; // 205
constexpr int WO = (W + 2 * PAD - KS) / PS + 1; // 205
constexpr int OB = PS + 1;        // 11 (with_padding=1 expands each block)
constexpr int OH = HO * OB;       // 2255
constexpr int OW = WO * OB;       // 2255
constexpr int NPIX = KS * KS;     // 121 window pixels, padded to 128 K-slots

// One wave32 per output tile.
//   D(16x16 f32) += A_hi(16x32 f16) x B(32x16 f16)   (4 K-chunks)
//   D(16x16 f32) += A_lo(16x32 f16) x B(32x16 f16)   (hi/lo split, ~2^-22 rel)
// A rows: 0=R, 1=G, 2=B, 3=valid(1/0), 4..15=0.  B[k][n] = one-hot(bin(pixel k)==n).
// K-slot s hosts window pixel (s/11, s%11); lane owns slots 4*lane..4*lane+3 so
// every LDS staging store is a vector op. K order is a permutation - sums don't care.
__global__ __launch_bounds__(32)
void pixel_effect_wmma(const float* __restrict__ rgb, float* __restrict__ out) {
    const int ox   = blockIdx.x;
    const int oy   = blockIdx.y;
    const int lane = threadIdx.x;          // 0..31 (wave32)

    // One-hot B operand, bin-major: sOH[n][k]. 4KB.
    __shared__ alignas(16) _Float16 sOH[NB][128];
    // A operand, hi and lo f16 planes: rows 0=R 1=G 2=B 3=valid 4=zeros.
    __shared__ alignas(16) _Float16 sAhi[5][128];   // 1.25KB
    __shared__ alignas(16) _Float16 sAlo[5][128];   // 1.25KB (rows 3,4 stay zero)

    // ---- Phase 0: vectorized zero-fill ----
    {
        const uint4 z = make_uint4(0, 0, 0, 0);
        uint4* ohp = reinterpret_cast<uint4*>(&sOH[0][0]);       // 4KB = 256 x 16B
        #pragma unroll
        for (int q = 0; q < 8; ++q) ohp[lane + 32 * q] = z;
        if (lane < 16)
            reinterpret_cast<uint4*>(&sAhi[4][0])[lane] = z;     // 256B zero row
        reinterpret_cast<uint4*>(&sAlo[3][0])[lane] = z;         // 512B rows 3..4
    }
    __syncthreads();

    // ---- Phase 1: stage 11x11 window (zero-padded at borders) into LDS ----
    const int by = oy * PS - PAD;
    const int bx = ox * PS - PAD;
    const size_t plane = (size_t)H * W;
    const int s0 = lane * 4;               // this lane's 4 consecutive K-slots
    int wy = s0 / KS, wx = s0 - wy * KS;   // one div/mod per lane; then +1 carries

    v4h rh4, gh4, bh4, vh4, rl4, gl4, bl4;
    int  bins[4]; bool valids[4];
    #pragma unroll
    for (int u = 0; u < 4; ++u) {
        const int s = s0 + u;
        float r = 0.f, g = 0.f, b = 0.f;
        int bin = 0; bool valid = false;
        if (s < NPIX) {
            const int gy = by + wy, gx = bx + wx;
            if (gy >= 0 && gy < H && gx >= 0 && gx < W) {
                const size_t p = (size_t)gy * W + gx;
                r = rgb[p];
                g = rgb[plane + p];
                b = rgb[2 * plane + p];
                const float mean = (r + g + b) / 3.0f;
                bin = (int)((mean / 256.0f) * (float)NB);   // matches reference op order
                bin = bin < 0 ? 0 : (bin > NB - 1 ? NB - 1 : bin);
                valid = true;
                // Warm L2 for the horizontally adjacent tile's window (global_prefetch_b8)
                if (gx + PS < W) __builtin_prefetch(&rgb[p + PS], 0, 1);
            }
        }
        const _Float16 hr = (_Float16)r, hg = (_Float16)g, hb = (_Float16)b;
        rh4[u] = hr; gh4[u] = hg; bh4[u] = hb;
        vh4[u] = valid ? (_Float16)1.0f : (_Float16)0.0f;
        rl4[u] = (_Float16)(r - (float)hr);
        gl4[u] = (_Float16)(g - (float)hg);
        bl4[u] = (_Float16)(b - (float)hb);
        bins[u] = bin; valids[u] = valid;
        wx += 1; if (wx == KS) { wx = 0; wy += 1; }   // next slot's pixel
    }
    // 7 vector LDS stores (ds_store_b64) instead of 28 scalar b16 stores
    *reinterpret_cast<v4h*>(&sAhi[0][s0]) = rh4;
    *reinterpret_cast<v4h*>(&sAhi[1][s0]) = gh4;
    *reinterpret_cast<v4h*>(&sAhi[2][s0]) = bh4;
    *reinterpret_cast<v4h*>(&sAhi[3][s0]) = vh4;
    *reinterpret_cast<v4h*>(&sAlo[0][s0]) = rl4;
    *reinterpret_cast<v4h*>(&sAlo[1][s0]) = gl4;
    *reinterpret_cast<v4h*>(&sAlo[2][s0]) = bl4;
    #pragma unroll
    for (int u = 0; u < 4; ++u)
        if (valids[u]) sOH[bins[u]][s0 + u] = (_Float16)1.0f;   // one-hot scatter
    __syncthreads();

    // ---- Phase 2: WMMA operands are pure aligned LDS loads (no VALU) ----
    const int gh  = lane >> 4;             // half-wave group (0/1)
    const int mn  = lane & 15;             // A-row / B-column for this lane
    const int rowHi = mn < 4 ? mn : 4;     // rows >= 4 read the zero row
    const int rowLo = mn < 3 ? mn : 3;     // rows >= 3 of lo-plane are zero
    const _Float16* ohRow = &sOH[mn][0];
    const _Float16* ahRow = &sAhi[rowHi][0];
    const _Float16* alRow = &sAlo[rowLo][0];

    v8f acc = {};
    #pragma unroll
    for (int t = 0; t < 4; ++t) {          // K chunks of 32 (covers 128 >= 121)
        const int kbase = 32 * t;

        // B 32x16 f16: lane group g holds K = g*16 + e -> 32B contiguous.
        const v8u bu = *reinterpret_cast<const v8u*>(&ohRow[kbase + gh * 16]);
        const v16h bm = __builtin_bit_cast(v16h, bu);

        // A 16x32 f16: K = kbase + g*16 + e (matches B)
        const uint4 h0 = *reinterpret_cast<const uint4*>(&ahRow[kbase + gh * 16]);
        const uint4 h1 = *reinterpret_cast<const uint4*>(&ahRow[kbase + gh * 16 + 8]);
        const uint4 l0 = *reinterpret_cast<const uint4*>(&alRow[kbase + gh * 16]);
        const uint4 l1 = *reinterpret_cast<const uint4*>(&alRow[kbase + gh * 16 + 8]);
        const v8u hu = { h0.x, h0.y, h0.z, h0.w, h1.x, h1.y, h1.z, h1.w };
        const v8u lu = { l0.x, l0.y, l0.z, l0.w, l1.x, l1.y, l1.z, l1.w };
        const v16h a_hi = __builtin_bit_cast(v16h, hu);
        const v16h a_lo = __builtin_bit_cast(v16h, lu);

        // (neg_a, A, neg_b, B, c_mod, C, reuse_a, reuse_b)
        acc = __builtin_amdgcn_wmma_f32_16x16x32_f16(false, a_hi, false, bm,
                                                     (short)0, acc, false, false);
        acc = __builtin_amdgcn_wmma_f32_16x16x32_f16(false, a_lo, false, bm,
                                                     (short)0, acc, false, false);
    }

    // ---- Phase 3: argmax over bins (counts = row M=3 -> acc[3], lanes 0..15) ----
    // key = count*16 + (15 - bin): max count wins, ties -> smallest bin (jnp.argmax).
    // counts are exact small integers in f32; lanes 16..31 carry row M=11 == 0.
    // xor-max reduction via ds_swizzle (group-of-32: offset = xor<<10 | and_mask 0x1f).
    int key = (((int)acc[3]) << 4) | (15 - mn);
    key = max(key, __builtin_amdgcn_ds_swizzle(key, 0x401F));  // xor 16
    key = max(key, __builtin_amdgcn_ds_swizzle(key, 0x201F));  // xor 8
    key = max(key, __builtin_amdgcn_ds_swizzle(key, 0x101F));  // xor 4
    key = max(key, __builtin_amdgcn_ds_swizzle(key, 0x081F));  // xor 2
    key = max(key, __builtin_amdgcn_ds_swizzle(key, 0x041F));  // xor 1
    const int   bsel = 15 - (key & 15);    // wave-uniform after reduction
    const float inv  = 1.0f / (float)(key >> 4);   // 1/cmax, shared by 3 channels

    // Broadcast D[M=c][N=bsel] from lane bsel (uniform index -> v_readlane)
    const float outv0 = __builtin_bit_cast(float,
        __builtin_amdgcn_readlane(__builtin_bit_cast(int, acc[0]), bsel)) * inv;
    const float outv1 = __builtin_bit_cast(float,
        __builtin_amdgcn_readlane(__builtin_bit_cast(int, acc[1]), bsel)) * inv;
    const float outv2 = __builtin_bit_cast(float,
        __builtin_amdgcn_readlane(__builtin_bit_cast(int, acc[2]), bsel)) * inv;

    // ---- Phase 4: write 3 x 11x11 block (10x10 value + zero pad row/col) ----
    const size_t oPlane = (size_t)OH * OW;
    const int oyB = oy * OB, oxB = ox * OB;
    int c = 0, i = lane / OB, j = lane % OB;   // flat idx = c*121 + i*11 + j
    #pragma unroll
    for (int q = 0; q < 12; ++q) {
        const int idx = lane + 32 * q;
        if (idx < 3 * OB * OB) {               // compile-time true for q < 11
            const float sel = (c == 0) ? outv0 : ((c == 1) ? outv1 : outv2);
            const float v = (i < PS && j < PS) ? sel : 0.0f;
            out[(size_t)c * oPlane + (size_t)(oyB + i) * OW + (oxB + j)] = v;
        }
        // advance flat index by 32 = +2 rows +10 cols (single carries)
        j += 10; i += 2;
        if (j >= OB) { j -= OB; i += 1; }
        if (i >= OB) { i -= OB; c += 1; }
    }
}

extern "C" void kernel_launch(void* const* d_in, const int* in_sizes, int n_in,
                              void* d_out, int out_size, void* d_ws, size_t ws_size,
                              hipStream_t stream) {
    (void)in_sizes; (void)n_in; (void)d_ws; (void)ws_size; (void)out_size;
    const float* rgb = (const float*)d_in[0];
    float* out = (float*)d_out;
    dim3 grid(WO, HO);   // 205 x 205 tiles, one wave32 workgroup each
    pixel_effect_wmma<<<grid, dim3(32, 1, 1), 0, stream>>>(rgb, out);
}
